// RelationNet_11209864642931
// MI455X (gfx1250) — matrix operand order repaired and checked by reading, not compiled, not run on hardware
//
#include <hip/hip_runtime.h>

typedef __attribute__((ext_vector_type(16))) __bf16 v16bf;
typedef __attribute__((ext_vector_type(8)))  float  v8f;
typedef __attribute__((ext_vector_type(8)))  __bf16 bf16x8;

// B=16, N=16, K=8, H=128 ; 2H=256 ; rows = B*N*K = 2048 ; pairs per (b,n1) = N*K*K = 1024

// ---------------------------------------------------------------------------
// Stage 1: L[r,o] = enc[r,:] . W1[o,0:128], R[r,o] = enc[r,:] . W1[o,128:256]
// r = (b,n,k) in [0,2048). Tiny (0.27 GFLOP) -> plain VALU, 8 rows per block
// to amortize W1 streaming.
// ---------------------------------------------------------------------------
__global__ __launch_bounds__(256) void relnet_stage1(
    const float* __restrict__ enc,   // [2048,128]
    const float* __restrict__ W1,    // [256,256]
    float* __restrict__ Lws,         // [2048,256]
    float* __restrict__ Rws)         // [2048,256]
{
    __shared__ float e[8][128];
    const int tid = threadIdx.x;
    const int r0  = blockIdx.x * 8;

    for (int i = tid; i < 8 * 128; i += 256)
        e[i >> 7][i & 127] = enc[r0 * 128 + i];
    __syncthreads();

    const int o = tid;
    float accL[8], accR[8];
#pragma unroll
    for (int j = 0; j < 8; ++j) { accL[j] = 0.f; accR[j] = 0.f; }

    const float* w1row = W1 + o * 256;
    for (int h = 0; h < 128; ++h) {
        const float wa = w1row[h];
        const float wb = w1row[128 + h];
#pragma unroll
        for (int j = 0; j < 8; ++j) {
            const float ev = e[j][h];
            accL[j] = fmaf(ev, wa, accL[j]);
            accR[j] = fmaf(ev, wb, accR[j]);
        }
    }
#pragma unroll
    for (int j = 0; j < 8; ++j) {
        Lws[(r0 + j) * 256 + o] = accL[j];
        Rws[(r0 + j) * 256 + o] = accR[j];
    }
}

// ---------------------------------------------------------------------------
// Stage 2: per (b,n1): h1 = relu(L[k1]+R[n2,k2]) [1024x256];
// h2 = relu(h1 @ W2^T); h3 = relu(h2 @ W3^T); out[b,n1,:] = colmean(h3).
// 256 threads = 8 waves; wave handles 16-row M-tiles.
// LDS (dynamic, 320KB): W2s 128KB | W3s 128KB | 8x8KB wave staging.
// Weights as bf16, 512B rows, 16B-granule XOR swizzle: granule ^= (row & 15).
// ---------------------------------------------------------------------------
__global__ __launch_bounds__(256) void relnet_main(
    const float* __restrict__ W2,
    const float* __restrict__ W3,
    const float* __restrict__ Lws,
    const float* __restrict__ Rws,
    float* __restrict__ out)
{
    extern __shared__ char smem[];
    char* const W2s   = smem;             // [256][512B]
    char* const W3s   = smem + 131072;    // [256][512B]
    char* const stag0 = smem + 262144;    // 8 waves * 8KB ([16][512B] each)

    const int tid   = threadIdx.x;
    const int wave  = tid >> 5;
    const int lane  = tid & 31;
    const int laneM = lane & 15;
    const int hi    = lane >> 4;          // 0: lanes 0-15, 1: lanes 16-31

    const int g  = blockIdx.x;            // (b, n1)
    const int b  = g >> 4;
    const int n1 = g & 15;

    // ---- fill W2s / W3s : f32 -> bf16, swizzled 16B granules ----
    for (int gr = tid; gr < 8192; gr += 256) {          // 256 rows * 32 granules
        const int row  = gr >> 5;
        const int gcol = gr & 31;
        const int i    = gcol << 3;
        const float* s2 = W2 + row * 256 + i;
        const float* s3 = W3 + row * 256 + i;
        bf16x8 v2, v3;
#pragma unroll
        for (int e2 = 0; e2 < 8; ++e2) { v2[e2] = (__bf16)s2[e2]; v3[e2] = (__bf16)s3[e2]; }
        const int addr = row * 512 + ((gcol ^ (row & 15)) << 4);
        *(bf16x8*)(W2s + addr) = v2;
        *(bf16x8*)(W3s + addr) = v3;
    }
    __syncthreads();

    char* const stag = stag0 + wave * 8192;
    const float* const Lbase = Lws + ((b * 16 + n1) * 8) * 256;   // + k1*256
    const float* const Rbase = Rws + (b * 128) * 256;             // + (n2*8+k2)*256

    float colacc[16];
#pragma unroll
    for (int i = 0; i < 16; ++i) colacc[i] = 0.f;

    for (int t = wave; t < 64; t += 8) {                // 16-row M-tiles
        const int m  = t * 16 + laneM;                  // pair row this lane owns
        const int n2 = m >> 6;
        const int k1 = (m >> 3) & 7;
        const int k2 = m & 7;
        const float* Lrow = Lbase + k1 * 256;
        const float* Rrow = Rbase + (n2 * 8 + k2) * 256;

        // ---- h1 A-fragments: 16-bit A layout, runs [32p+8*hi,+8) and +16 ----
        v16bf afrag[8];
#pragma unroll
        for (int p = 0; p < 8; ++p) {
            const int i0 = p * 32 + hi * 8;
#pragma unroll
            for (int rs = 0; rs < 2; ++rs) {
                const int ib = i0 + rs * 16;
                const float4 l0 = *(const float4*)(Lrow + ib);
                const float4 l1 = *(const float4*)(Lrow + ib + 4);
                const float4 r0 = *(const float4*)(Rrow + ib);
                const float4 r1 = *(const float4*)(Rrow + ib + 4);
                float s[8] = { l0.x + r0.x, l0.y + r0.y, l0.z + r0.z, l0.w + r0.w,
                               l1.x + r1.x, l1.y + r1.y, l1.z + r1.z, l1.w + r1.w };
#pragma unroll
                for (int e2 = 0; e2 < 8; ++e2) {
                    const float v = s[e2] > 0.f ? s[e2] : 0.f;  // ReLU (layer 1)
                    afrag[p][rs * 8 + e2] = (__bf16)v;
                }
            }
        }

        // ---- layer 2: 16 output tiles, 8 wmma each; relu->bf16->staging ----
#pragma unroll 1
        for (int ot = 0; ot < 16; ++ot) {
            v8f acc = {};
            const int orow = ot * 16 + laneM;           // W2 row = output col
#pragma unroll
            for (int p = 0; p < 8; ++p) {
                const int g0 = p * 4 + hi;              // K-run granules
                const bf16x8 blo = *(const bf16x8*)(W2s + orow * 512 + ((g0 ^ laneM) << 4));
                const bf16x8 bhi = *(const bf16x8*)(W2s + orow * 512 + (((g0 + 2) ^ laneM) << 4));
                v16bf bfrag;
#pragma unroll
                for (int e2 = 0; e2 < 8; ++e2) { bfrag[e2] = blo[e2]; bfrag[8 + e2] = bhi[e2]; }
                acc = __builtin_amdgcn_wmma_f32_16x16x32_bf16(
                          false, afrag[p], false, bfrag, (short)0, acc, false, false);
            }
            // D layout: lane holds col orow, rows hi*8+r -> staging [row][col]
#pragma unroll
            for (int r = 0; r < 8; ++r) {
                const float v = acc[r] > 0.f ? acc[r] : 0.f;
                const int mr  = hi * 8 + r;
                const int gsw = ((orow >> 3) ^ mr);
                *(__bf16*)(stag + mr * 512 + (gsw << 4) + (orow & 7) * 2) = (__bf16)v;
            }
        }

        // ---- layer 3 A-fragments from wave-private staging ----
        v16bf a2[8];
#pragma unroll
        for (int p = 0; p < 8; ++p) {
            const int g0 = p * 4 + hi;                  // (32p+8hi)>>3
            const bf16x8 lo  = *(const bf16x8*)(stag + laneM * 512 + ((g0 ^ laneM) << 4));
            const bf16x8 hi8 = *(const bf16x8*)(stag + laneM * 512 + (((g0 + 2) ^ laneM) << 4));
#pragma unroll
            for (int e2 = 0; e2 < 8; ++e2) { a2[p][e2] = lo[e2]; a2[p][8 + e2] = hi8[e2]; }
        }

        // ---- layer 3 + column partial sums (mean accumulation) ----
#pragma unroll 1
        for (int ot = 0; ot < 16; ++ot) {
            v8f acc = {};
            const int orow = ot * 16 + laneM;
#pragma unroll
            for (int p = 0; p < 8; ++p) {
                const int g0 = p * 4 + hi;
                const bf16x8 blo = *(const bf16x8*)(W3s + orow * 512 + ((g0 ^ laneM) << 4));
                const bf16x8 bhi = *(const bf16x8*)(W3s + orow * 512 + (((g0 + 2) ^ laneM) << 4));
                v16bf bfrag;
#pragma unroll
                for (int e2 = 0; e2 < 8; ++e2) { bfrag[e2] = blo[e2]; bfrag[8 + e2] = bhi[e2]; }
                acc = __builtin_amdgcn_wmma_f32_16x16x32_bf16(
                          false, a2[p], false, bfrag, (short)0, acc, false, false);
            }
            float s = 0.f;
#pragma unroll
            for (int r = 0; r < 8; ++r) { const float v = acc[r] > 0.f ? acc[r] : 0.f; s += v; }
            colacc[ot] += s;
        }
    }

    // merge rows 0-7 with 8-15 (lane L with L^16 hold the same column)
#pragma unroll
    for (int ot = 0; ot < 16; ++ot)
        colacc[ot] += __shfl_xor(colacc[ot], 16, 32);

    // cross-wave reduction through (now reusable) staging
    if (lane < 16) {
#pragma unroll
        for (int ot = 0; ot < 16; ++ot)
            *(float*)(stag + (ot * 16 + laneM) * 4) = colacc[ot];
    }
    __syncthreads();
    {
        float s = 0.f;
#pragma unroll
        for (int w = 0; w < 8; ++w)
            s += *(const float*)(stag0 + w * 8192 + tid * 4);
        out[g * 256 + tid] = s * (1.0f / 1024.0f);      // mean over N*K*K = 1024
    }
}

// ---------------------------------------------------------------------------
extern "C" void kernel_launch(void* const* d_in, const int* in_sizes, int n_in,
                              void* d_out, int out_size, void* d_ws, size_t ws_size,
                              hipStream_t stream) {
    const float* enc = (const float*)d_in[0];   // [16,16,8,128]
    const float* W1  = (const float*)d_in[1];   // [256,256]
    const float* W2  = (const float*)d_in[2];   // [256,256]
    const float* W3  = (const float*)d_in[3];   // [256,256]
    float* out = (float*)d_out;                 // [16,16,256]

    float* Lws = (float*)d_ws;                  // [2048,256] f32  (2 MB)
    float* Rws = Lws + 2048 * 256;              // [2048,256] f32  (2 MB)

    relnet_stage1<<<256, 256, 0, stream>>>(enc, W1, Lws, Rws);
    relnet_main<<<256, 256, 327680, stream>>>(W2, W3, Lws, Rws, out);
}
